// LinearAttention_2430951489569
// MI455X (gfx1250) — matrix operand order repaired and checked by reading, not compiled, not run on hardware
//
#include <hip/hip_runtime.h>
#include <hip/hip_bf16.h>
#include <stdint.h>

#define BB     16
#define DIM    256
#define NPIX   4096
#define HEADS  4
#define DHEAD  32
#define HIDDEN 128
#define MEM    4
#define OUT3   384
#define TILE_N 64

typedef __attribute__((ext_vector_type(16))) _Float16 v16h;
typedef __attribute__((ext_vector_type(8)))  float    v8f;
typedef __attribute__((ext_vector_type(4)))  uint32_t u32x4;
typedef __attribute__((ext_vector_type(8)))  int      i32x8;
typedef __attribute__((ext_vector_type(4)))  int      i32x4;

union Frag {
    v16h     v;
    uint32_t u[8];
    _Float16 h[16];
};

__device__ __forceinline__ v8f wmma_f16(const Frag& a, const Frag& b, v8f c) {
    return __builtin_amdgcn_wmma_f32_16x16x32_f16(false, a.v, false, b.v,
                                                  (short)0, c, false, false);
}

__device__ __forceinline__ uint32_t pack2(_Float16 h0, _Float16 h1) {
    uint32_t pk;
    ((_Float16*)&pk)[0] = h0;
    ((_Float16*)&pk)[1] = h1;
    return pk;
}

// ---------------------------------------------------------------- kernel 1
__global__ void k_convert_w(const float* __restrict__ wqkv,
                            const float* __restrict__ wout,
                            _Float16* __restrict__ wqkv_h,
                            _Float16* __restrict__ wout_h) {
    int i = blockIdx.x * blockDim.x + threadIdx.x;
    if (i < OUT3 * DIM) wqkv_h[i] = (_Float16)wqkv[i];
    int j = i - OUT3 * DIM;
    if (j >= 0 && j < DIM * HIDDEN) wout_h[j] = (_Float16)wout[j];
}

// ---------------------------------------------------------------- kernel 2
// RMS-norm + QKV GEMM (384x256 * 256x64) + q-softmax, per (b, 64-pixel tile).
// x tile staged via the Tensor Data Mover into LDS with hardware stride-65
// padding (pad 1 dword after every 64 dwords), aliasing the outT region.
__global__ __launch_bounds__(256)
void k_rmsnorm_qkv(const float* __restrict__ x,
                   const float* __restrict__ g1,
                   const _Float16* __restrict__ wqkv_h,
                   _Float16* __restrict__ q_soft,
                   _Float16* __restrict__ k_raw,
                   _Float16* __restrict__ v_h) {
    const int b    = blockIdx.y;
    const int n0   = blockIdx.x * TILE_N;
    const int t    = threadIdx.x;
    const int lane = t & 31, wave = t >> 5;
    const int half = lane >> 4;

    __shared__ _Float16 xnT[TILE_N][DIM + 2];   // [col][c], stride 258 (odd dwords)
    __shared__ float    outT[OUT3 * 65];        // GEMM C [row][col]; aliased as x stage
    __shared__ float    invr[TILE_N];
    __shared__ float    red[256];
    __shared__ float    qmax[HEADS][TILE_N];
    __shared__ float    qscl[HEADS][TILE_N];

    const float* xb = x + (size_t)b * DIM * NPIX;
    float* xstage = outT;                       // [c][col], stride 65 (16639 <= 24960)

#if __has_include(<hip/amd_detail/amd_gfx1250_TDM.h>) || \
    __has_builtin(__builtin_amdgcn_tensor_load_to_lds)
    // --- TDM path: 2D tile 256x64 f32, row stride NPIX, LDS padding of
    // 1 dword per 64 dwords -> [c][col] stride-65 layout written by hardware.
    if (wave == 0) {
        uint64_t gaddr = (uint64_t)(uintptr_t)(xb + n0);
        uint32_t laddr = (uint32_t)(uintptr_t)(void*)xstage;
        u32x4 g0;
        g0.x = 1u;                                   // count = 1 valid descriptor
        g0.y = laddr;                                // lds_addr (bytes)
        g0.z = (uint32_t)gaddr;                      // global_addr[31:0]
        g0.w = (uint32_t)((gaddr >> 32) & 0x1FFFFFFu) | (2u << 30);  // [56:32]+type=2
        i32x8 g1d = {
            (int)((2u << 16) | (1u << 20) | (5u << 22)),  // data_size=4B, pad_en,
                                                          // pad_interval=64dw, pad=1dw
            (int)(((uint32_t)NPIX & 0xFFFFu) << 16),      // tensor_dim0[15:0]=4096
            (int)(((uint32_t)DIM  & 0xFFFFu) << 16),      // dim0 hi=0, tensor_dim1=256
            (int)((uint32_t)TILE_N << 16),                // dim1 hi=0, tile_dim0=64
            (int)DIM,                                     // tile_dim1=256, tile_dim2=0
            (int)NPIX,                                    // tensor_dim0_stride lo
            0, 0                                          // stride hi, dim1_stride
        };
        i32x4 z4 = {0, 0, 0, 0};
#if __has_include(<hip/amd_detail/amd_gfx1250_TDM.h>)
        // amdgpu-toolchain (clang-23 / therock-10): 6-arg builtin
        i32x8 z8 = {0, 0, 0, 0, 0, 0, 0, 0};
        __builtin_amdgcn_tensor_load_to_lds(g0, g1d, z4, z4, z8, 0);
#else
        // ROCm 7.2 (clang-22): 5-arg builtin
        __builtin_amdgcn_tensor_load_to_lds(g0, g1d, z4, z4, 0);
#endif
        __builtin_amdgcn_s_wait_tensorcnt(0);
    }
#else
    // --- fallback: coalesced VGPR staging
    for (int i = 0; i < DIM * TILE_N / 256; ++i) {   // 64 iters
        int idx = i * 256 + t;
        int c = idx >> 6, col = idx & 63;
        xstage[c * 65 + col] = xb[(size_t)c * NPIX + n0 + col];
    }
#endif
    __syncthreads();

    // per-column ||x|| from LDS
    {
        int col = t >> 2, sub = t & 3;
        float s = 0.f;
        for (int cc = 0; cc < 64; ++cc) {
            float v = xstage[(sub * 64 + cc) * 65 + col];
            s += v * v;
        }
        red[t] = s;
        __syncthreads();
        if (sub == 0) {
            float n = sqrtf(red[t] + red[t + 1] + red[t + 2] + red[t + 3]);
            invr[col] = 16.0f / fmaxf(n, 1e-12f);   // sqrt(256)/||x||
        }
        __syncthreads();
    }
    // normalized f16 tile, transposed [col][c], packed dword LDS stores
    {
        int col = t & 63, cg = t >> 6;
        float iv = invr[col];
        for (int cc = 0; cc < 64; cc += 2) {
            int c = cg * 64 + cc;
            _Float16 h0 = (_Float16)(xstage[c * 65 + col] * iv * g1[c]);
            _Float16 h1 = (_Float16)(xstage[(c + 1) * 65 + col] * iv * g1[c + 1]);
            *(uint32_t*)&xnT[col][c] = pack2(h0, h1);
        }
    }
    __syncthreads();                 // xstage dead from here on; outT reusable

    // WMMA GEMM: wave w owns m-tiles 3w..3w+2, all 4 n-tiles
    v8f acc[3][4] = {};
    for (int kk = 0; kk < DIM / 32; ++kk) {
        Frag a[3], bf[4];
        for (int i = 0; i < 3; ++i) {
            int m = (wave * 3 + i) * 16 + (lane & 15);
            const _Float16* wr = wqkv_h + (size_t)m * DIM + kk * 32;
            for (int r = 0; r < 8; ++r) {
                int kb = half * 16 + 2 * r;
                a[i].u[r] = *(const uint32_t*)(wr + kb);
            }
        }
        int kbase = (lane < 16) ? 0 : 16;
        for (int j = 0; j < 4; ++j) {
            int col = j * 16 + (lane & 15);
            const _Float16* sr = &xnT[col][kk * 32 + kbase];
            for (int r = 0; r < 8; ++r)
                bf[j].u[r] = *(const uint32_t*)(sr + 2 * r);
        }
        for (int i = 0; i < 3; ++i)
            for (int j = 0; j < 4; ++j)
                acc[i][j] = wmma_f16(a[i], bf[j], acc[i][j]);
    }
    // spill C tiles to LDS (C layout: VGPR r -> M = r + 8*(lane>=16), N = lane%16)
    for (int i = 0; i < 3; ++i) {
        int mt = wave * 3 + i;
        for (int j = 0; j < 4; ++j)
            for (int r = 0; r < 8; ++r) {
                int m   = mt * 16 + r + half * 8;
                int col = j * 16 + (lane & 15);
                outT[m * 65 + col] = acc[i][j][r];
            }
    }
    __syncthreads();

    // q softmax over DHEAD per (head, col)
    {
        int head = t >> 6, col = t & 63;
        float mx = -1e30f;
        for (int d = 0; d < DHEAD; ++d)
            mx = fmaxf(mx, outT[(head * DHEAD + d) * 65 + col]);
        float den = 0.f;
        for (int d = 0; d < DHEAD; ++d)
            den += __expf(outT[(head * DHEAD + d) * 65 + col] - mx);
        qmax[head][col] = mx;
        qscl[head][col] = 0.17677669529663687f / den;   // DHEAD^-0.5 / den
    }
    __syncthreads();

    // packed f16x2 stores of q (softmaxed), k, v
    {
        int cp = t & 31, rg = t >> 5;
        int c0 = cp * 2;
        for (int it = 0; it < OUT3 / 8; ++it) {     // 48 iters
            int r = it * 8 + rg;
            float v0 = outT[r * 65 + c0];
            float v1 = outT[r * 65 + c0 + 1];
            _Float16 h0, h1;
            _Float16* dst;
            if (r < HIDDEN) {
                int head = r >> 5;
                h0 = (_Float16)(__expf(v0 - qmax[head][c0]) * qscl[head][c0]);
                h1 = (_Float16)(__expf(v1 - qmax[head][c0 + 1]) * qscl[head][c0 + 1]);
                dst = q_soft + (size_t)(b * HIDDEN + r) * NPIX + n0 + c0;
            } else if (r < 2 * HIDDEN) {
                h0 = (_Float16)v0; h1 = (_Float16)v1;
                dst = k_raw + (size_t)(b * HIDDEN + r - HIDDEN) * NPIX + n0 + c0;
            } else {
                h0 = (_Float16)v0; h1 = (_Float16)v1;
                dst = v_h + (size_t)(b * HIDDEN + r - 2 * HIDDEN) * NPIX + n0 + c0;
            }
            *(uint32_t*)dst = pack2(h0, h1);
        }
    }
}

// ---------------------------------------------------------------- kernel 3
// per (b,h): k-softmax stats (LDS-resident) + ctx = exp(k-max).v^T / ksum, WMMA K=4096
__global__ __launch_bounds__(256)
void k_context(const _Float16* __restrict__ k_raw,
               const _Float16* __restrict__ v_h,
               const float* __restrict__ mem_kv,
               float* __restrict__ ctx) {
    const int h = blockIdx.x, b = blockIdx.y;
    const int t = threadIdx.x, lane = t & 31, wave = t >> 5;
    const int half = lane >> 4;
    const size_t rowbase = (size_t)(b * HIDDEN + h * DHEAD) * NPIX;
    const int sbase = (b * HEADS + h) * DHEAD;

    __shared__ float part[8 * DHEAD * DHEAD];
    __shared__ float red[256];
    __shared__ float kmaxS[DHEAD];
    __shared__ float ksumS[DHEAD];

    const float* mk = mem_kv + (size_t)(h * DHEAD) * MEM;
    const float* mv = mem_kv + (size_t)(HEADS * DHEAD + h * DHEAD) * MEM;

    // phase 0: per-d max and sum(exp) over 4096 k values + mem tokens
    {
        int d = t >> 3, sub = t & 7;
        const uint32_t* kr =
            (const uint32_t*)(k_raw + rowbase + (size_t)d * NPIX) + sub * 256;
        float mx = -1e30f;
        for (int i = 0; i < 256; ++i) {
            uint32_t raw = kr[i];
            mx = fmaxf(mx, (float)((const _Float16*)&raw)[0]);
            mx = fmaxf(mx, (float)((const _Float16*)&raw)[1]);
        }
        red[t] = mx;
        __syncthreads();
        if (sub == 0) {
            float m = -1e30f;
            for (int i = 0; i < 8; ++i) m = fmaxf(m, red[t + i]);
            for (int j = 0; j < MEM; ++j) m = fmaxf(m, mk[d * MEM + j]);
            kmaxS[d] = m;
        }
        __syncthreads();
        float kmx = kmaxS[d];
        float s = 0.f;
        for (int i = 0; i < 256; ++i) {
            uint32_t raw = kr[i];
            s += __expf((float)((const _Float16*)&raw)[0] - kmx);
            s += __expf((float)((const _Float16*)&raw)[1] - kmx);
        }
        red[t] = s;
        __syncthreads();
        if (sub == 0) {
            float ss = 0.f;
            for (int i = 0; i < 8; ++i) ss += red[t + i];
            for (int j = 0; j < MEM; ++j) ss += __expf(mk[d * MEM + j] - kmaxS[d]);
            ksumS[d] = ss;
        }
        __syncthreads();
    }

    // phase 1: WMMA over K=4096, 2x2 tiles of 16x16, wave w owns n in [512w,512w+512)
    v8f acc[2][2] = {};
    float kmx0 = kmaxS[lane & 15];
    float kmx1 = kmaxS[16 + (lane & 15)];
    for (int s = 0; s < 16; ++s) {
        int nb = wave * 512 + s * 32;
        Frag a[2], bf[2];
        for (int i = 0; i < 2; ++i) {
            int d = i * 16 + (lane & 15);
            float kmx = i ? kmx1 : kmx0;
            const _Float16* kr = k_raw + rowbase + (size_t)d * NPIX + nb;
            for (int r = 0; r < 8; ++r) {
                int kb = half * 16 + 2 * r;
                uint32_t raw = *(const uint32_t*)(kr + kb);
                _Float16 lo = ((const _Float16*)&raw)[0];
                _Float16 hi = ((const _Float16*)&raw)[1];
                a[i].h[2 * r]     = (_Float16)__expf((float)lo - kmx);
                a[i].h[2 * r + 1] = (_Float16)__expf((float)hi - kmx);
            }
        }
        int kbase = (lane < 16) ? 0 : 16;
        for (int j = 0; j < 2; ++j) {
            int e = j * 16 + (lane & 15);
            const _Float16* vr = v_h + rowbase + (size_t)e * NPIX + nb;
            for (int r = 0; r < 8; ++r)
                bf[j].u[r] = *(const uint32_t*)(vr + kbase + 2 * r);
        }
        for (int i = 0; i < 2; ++i)
            for (int j = 0; j < 2; ++j)
                acc[i][j] = wmma_f16(a[i], bf[j], acc[i][j]);
    }
    for (int i = 0; i < 2; ++i)
        for (int j = 0; j < 2; ++j)
            for (int r = 0; r < 8; ++r) {
                int d = i * 16 + r + half * 8;
                int e = j * 16 + (lane & 15);
                part[wave * 1024 + d * 32 + e] = acc[i][j][r];
            }
    __syncthreads();

    for (int q = 0; q < 4; ++q) {
        int idx = t * 4 + q;
        int d = idx >> 5, e = idx & 31;
        float s = 0.f;
        for (int w = 0; w < 8; ++w) s += part[w * 1024 + idx];
        float mx = kmaxS[d];
        for (int m = 0; m < MEM; ++m)
            s += __expf(mk[d * MEM + m] - mx) * mv[e * MEM + m];
        ctx[(size_t)(sbase + d) * DHEAD + e] = s / ksumS[d];
    }
}

// ---------------------------------------------------------------- kernel 4
// hid = ctx^T x q (per head), y = W_out x hid + b, final RMS-norm
__global__ __launch_bounds__(256)
void k_attn_out(const _Float16* __restrict__ q_soft,
                const float* __restrict__ ctx,
                const _Float16* __restrict__ wout_h,
                const float* __restrict__ b_out,
                const float* __restrict__ g2,
                float* __restrict__ out) {
    const int b  = blockIdx.y;
    const int n0 = blockIdx.x * TILE_N;
    const int t = threadIdx.x, lane = t & 31, wave = t >> 5;
    const int half = lane >> 4;

    __shared__ _Float16 qT[TILE_N][HIDDEN + 2];    // [col][d], stride 130
    __shared__ _Float16 hidT[TILE_N][HIDDEN + 2];  // [col][k], stride 130
    __shared__ float    ytile[DIM * 65];
    __shared__ float    red[256];
    __shared__ float    invr[TILE_N];

    // stage q tile: coalesced global reads, LDS transpose
    for (int i = 0; i < HIDDEN * TILE_N / 256; ++i) {   // 32 iters
        int idx = i * 256 + t;
        int row = idx >> 6, col = idx & 63;
        qT[col][row] = q_soft[(size_t)(b * HIDDEN + row) * NPIX + n0 + col];
    }
    __syncthreads();

    // stage A: wave w -> hidden rows [16w,16w+16), K = DHEAD = 32 (one WMMA step)
    {
        int head = wave >> 1;
        int eh   = (wave & 1) * 16 + (lane & 15);
        const float* cb = ctx + (size_t)(b * HEADS + head) * DHEAD * DHEAD;
        Frag a;
        for (int r = 0; r < 8; ++r) {
            int kb = half * 16 + 2 * r;
            a.h[2 * r]     = (_Float16)cb[(size_t)kb * DHEAD + eh];
            a.h[2 * r + 1] = (_Float16)cb[(size_t)(kb + 1) * DHEAD + eh];
        }
        int kbase = (lane < 16) ? 0 : 16;
        for (int j = 0; j < 4; ++j) {
            int col = j * 16 + (lane & 15);
            const _Float16* qr = &qT[col][head * DHEAD + kbase];
            Frag bf;
            for (int r = 0; r < 8; ++r)
                bf.u[r] = *(const uint32_t*)(qr + 2 * r);
            v8f c = {};
            c = wmma_f16(a, bf, c);
            for (int r = 0; r < 8; ++r) {
                int m = wave * 16 + r + half * 8;
                hidT[col][m] = (_Float16)c[r];
            }
        }
    }
    __syncthreads();

    // stage C: 256x128 * 128x64 projection
    v8f acc[2][4] = {};
    for (int kk = 0; kk < HIDDEN / 32; ++kk) {
        Frag a[2], bf[4];
        for (int i = 0; i < 2; ++i) {
            int m = (wave * 2 + i) * 16 + (lane & 15);
            const _Float16* wr = wout_h + (size_t)m * HIDDEN + kk * 32;
            for (int r = 0; r < 8; ++r) {
                int kb = half * 16 + 2 * r;
                a[i].u[r] = *(const uint32_t*)(wr + kb);
            }
        }
        int kbase = (lane < 16) ? 0 : 16;
        for (int j = 0; j < 4; ++j) {
            int col = j * 16 + (lane & 15);
            const _Float16* hr = &hidT[col][kk * 32 + kbase];
            for (int r = 0; r < 8; ++r)
                bf[j].u[r] = *(const uint32_t*)(hr + 2 * r);
        }
        for (int i = 0; i < 2; ++i)
            for (int j = 0; j < 4; ++j)
                acc[i][j] = wmma_f16(a[i], bf[j], acc[i][j]);
    }
    for (int i = 0; i < 2; ++i)
        for (int j = 0; j < 4; ++j)
            for (int r = 0; r < 8; ++r) {
                int m   = (wave * 2 + i) * 16 + r + half * 8;
                int col = j * 16 + (lane & 15);
                ytile[m * 65 + col] = acc[i][j][r] + b_out[m];
            }
    __syncthreads();

    // final column RMS-norm
    {
        int col = t >> 2, sub = t & 3;
        float s = 0.f;
        for (int rr = 0; rr < 64; ++rr) {
            float v = ytile[(sub * 64 + rr) * 65 + col];
            s += v * v;
        }
        red[t] = s; __syncthreads();
        if (sub == 0) {
            float n = sqrtf(red[t] + red[t + 1] + red[t + 2] + red[t + 3]);
            invr[col] = 16.0f / fmaxf(n, 1e-12f);
        }
        __syncthreads();
    }
    {
        int col = t & 63, rg = t >> 6;
        float iv = invr[col];
        float* ob = out + (size_t)b * DIM * NPIX + n0 + col;
        for (int rr = 0; rr < 64; ++rr) {
            int r = rg * 64 + rr;
            ob[(size_t)r * NPIX] = ytile[r * 65 + col] * iv * g2[r];
        }
    }
}

// ---------------------------------------------------------------- launch
extern "C" void kernel_launch(void* const* d_in, const int* in_sizes, int n_in,
                              void* d_out, int out_size, void* d_ws, size_t ws_size,
                              hipStream_t stream) {
    (void)in_sizes; (void)n_in; (void)out_size; (void)ws_size;
    const float* x     = (const float*)d_in[0];
    const float* g1    = (const float*)d_in[1];
    const float* wqkv  = (const float*)d_in[2];
    const float* memkv = (const float*)d_in[3];
    const float* wout  = (const float*)d_in[4];
    const float* bout  = (const float*)d_in[5];
    const float* g2    = (const float*)d_in[6];
    float* out = (float*)d_out;

    char* ws = (char*)d_ws;
    size_t off = 0;
    _Float16* wqkv_h = (_Float16*)(ws + off); off += (size_t)OUT3 * DIM * 2;
    _Float16* wout_h = (_Float16*)(ws + off); off += (size_t)DIM * HIDDEN * 2;
    _Float16* q_soft = (_Float16*)(ws + off); off += (size_t)BB * HIDDEN * NPIX * 2;
    _Float16* k_raw  = (_Float16*)(ws + off); off += (size_t)BB * HIDDEN * NPIX * 2;
    _Float16* v_h    = (_Float16*)(ws + off); off += (size_t)BB * HIDDEN * NPIX * 2;
    float* ctx  = (float*)(ws + off); off += (size_t)BB * HEADS * DHEAD * DHEAD * 4;

    k_convert_w<<<512, 256, 0, stream>>>(wqkv, wout, wqkv_h, wout_h);
    k_rmsnorm_qkv<<<dim3(NPIX / TILE_N, BB), 256, 0, stream>>>(x, g1, wqkv_h,
                                                               q_soft, k_raw, v_h);
    k_context<<<dim3(HEADS, BB), 256, 0, stream>>>(k_raw, v_h, memkv, ctx);
    k_attn_out<<<dim3(NPIX / TILE_N, BB), 256, 0, stream>>>(q_soft, ctx, wout_h,
                                                            bout, g2, out);
}
